// TGAR_73452530696953
// MI455X (gfx1250) — matrix operand order repaired and checked by reading, not compiled, not run on hardware
//
#include <hip/hip_runtime.h>
#include <cstdint>

// ---------------------------------------------------------------------------
// Problem constants (from reference)
// ---------------------------------------------------------------------------
#define N_NODES 50000
#define N_EDGES 800000
#define HIDDEN  128
#define NHEAD   8
#define HDIM    16

#define CDIV(a, b) (((a) + (b) - 1) / (b))

typedef __attribute__((ext_vector_type(16))) _Float16 v16h;
typedef __attribute__((ext_vector_type(8)))  float    v8f;

union HalfPack { unsigned int u; _Float16 h[2]; };
union FragA    { v16h h; unsigned int u[8]; };
union FragB    { v16h h; uint4 q[2]; };

__device__ __forceinline__ unsigned int packf2(float a, float b) {
  HalfPack p;
  p.h[0] = (_Float16)a;
  p.h[1] = (_Float16)b;
  return p.u;
}

// ---------------------------------------------------------------------------
// C[M,128] = act( A[M,128] @ W[128,128]^T + bias )   (row-major, K = 128)
// One block = 256 threads = 8 waves; block covers 128 rows of A.
// Each wave: one 16-row M tile x all eight 16-col N tiles,
// 4x v_wmma_f32_16x16x32_f16 per N tile (K = 128).
//
// A: LDS row-major packed-f16 dwords with XOR dword swizzle (conflict-free
//    fragment gathers; float4-coalesced HBM reads for the streamed operand).
// W: LDS *fragment-major* [frag(nt,kc)][half][lane][4dw] so each B operand
//    is two contiguous 16B chunks per lane -> two ds_load_b128 directly into
//    the contiguous even-aligned VGPR block WMMA needs (no repacking moves).
//    Lane stride 16B = 2-pass, bank-conflict-free.  Staging stores are
//    lane-linear (conflict-free); W is 64KB and L2-resident, so the 16B-
//    granular global reads during staging are free.
// ---------------------------------------------------------------------------
template <int ACT>  // 0 = none, 1 = relu
__global__ __launch_bounds__(256)
void gemm128_wmma(const float* __restrict__ A, const float* __restrict__ W,
                  const float* __restrict__ bias, float* __restrict__ C, int M) {
  __shared__ unsigned int As[128 * 64];      // 32 KB, row-major + swizzle
  __shared__ uint4        WsF[32 * 2 * 32];  // 32 KB, fragment-major
  unsigned int* Wsw = (unsigned int*)WsF;
  const int tid  = threadIdx.x;
  const int row0 = blockIdx.x * 128;

  // ---- Stage A (float4, zero-padded tail) as packed-f16 dwords ----
#pragma unroll
  for (int i = 0; i < 16; ++i) {
    int idx = tid + i * 256;        // 0..4095
    int r   = idx >> 5;             // 0..127
    int c4  = (idx & 31) << 2;      // 0,4,...,124
    int dw  = c4 >> 1;              // dword column base (even)
    int sw  = r & 15;
    float4 av = make_float4(0.f, 0.f, 0.f, 0.f);
    if (row0 + r < M) av = *(const float4*)(A + (size_t)(row0 + r) * 128 + c4);
    As[r * 64 + ((dw + 0) ^ sw)] = packf2(av.x, av.y);
    As[r * 64 + ((dw + 1) ^ sw)] = packf2(av.z, av.w);
  }

  // ---- Stage W fragment-major ----
  // B fragment (32x16 f16): lane = 16*hi + (n&15); dword v holds W[n][k],
  // W[n][k+1] with k = kc*32 + (v>>2)*16 + hi*8 + 2*(v&3).  Fragment id fb = nt*4 + kc.
  // Position: Wsw[(((fb*2 + (v>>2))*32 + lane)*4 + (v&3)].
#pragma unroll
  for (int j = 0; j < 16; ++j) {
    int p2  = j * 256 + tid;             // dword-pair index 0..4095
    int d0  = (p2 << 1) & 3;             // 0 or 2
    int lf  = (p2 >> 1) & 31;            // fragment lane
    int h   = (p2 >> 6) & 1;             // operand half (v>>2)
    int fb  = p2 >> 7;                   // 0..31 = nt*4 + kc
    int n   = ((fb >> 2) << 4) + (lf & 15);
    int dwc = ((fb & 3) << 4) + (h << 3) + ((lf >> 4) << 2) + d0;  // dword col
    const float* wp = W + (size_t)n * 128 + (dwc << 1);            // 4B-safe
    int base = ((fb * 2 + h) * 32 + lf) * 4 + d0;
    Wsw[base + 0] = packf2(wp[0], wp[1]);
    Wsw[base + 1] = packf2(wp[2], wp[3]);
  }
  __syncthreads();

  const int wave = tid >> 5;
  const int lane = tid & 31;
  const int hi   = lane >> 4;   // half-wave select
  const int l15  = lane & 15;

  const int ar  = wave * 16 + l15;  // A row for this lane
  const int asw = ar & 15;

  v8f acc[8] = {};
#pragma unroll
  for (int kc = 0; kc < 4; ++kc) {
    // A fragment (16x32 f16, ISA 7.12.2): dword v holds K pair; lanes0-15
    // K base 0, lanes16-31 K base 8; dwords 0-3 -> base+v, 4-7 -> base+4+v.
    FragA a;
    const int kc2 = kc * 16;
#pragma unroll
    for (int v = 0; v < 8; ++v) {
      int dw = kc2 + hi * 4 + ((v < 4) ? v : 4 + v);
      a.u[v] = As[ar * 64 + (dw ^ asw)];
    }
#pragma unroll
    for (int nt = 0; nt < 8; ++nt) {
      FragB b;
      const int fb = nt * 4 + kc;
      b.q[0] = WsF[(fb * 2 + 0) * 32 + lane];
      b.q[1] = WsF[(fb * 2 + 1) * 32 + lane];
      acc[nt] = __builtin_amdgcn_wmma_f32_16x16x32_f16(
          false, a.h, false, b.h, (short)0, acc[nt], false, false);
    }
  }

  // D layout: vgpr r -> row r (lanes0-15) / row 8+r (lanes16-31), col = lane&15.
  const int rbase = row0 + wave * 16 + hi * 8;
#pragma unroll
  for (int nt = 0; nt < 8; ++nt) {
    int col  = nt * 16 + l15;
    float bv = bias ? bias[col] : 0.0f;
#pragma unroll
    for (int r = 0; r < 8; ++r) {
      int row = rbase + r;
      if (row < M) {
        float val = acc[nt][r] + bv;
        if (ACT == 1) val = fmaxf(val, 0.0f);
        C[(size_t)row * 128 + col] = val;
      }
    }
  }
}

// ---------------------------------------------------------------------------
// Elementwise helpers
// ---------------------------------------------------------------------------
__global__ void k_fill0(float* p, int n) {
  int i = blockIdx.x * 256 + threadIdx.x;
  if (i < n) p[i] = 0.f;
}
__global__ void k_mul(const float* a, const float* b, float* o, int n) {
  int i = blockIdx.x * 256 + threadIdx.x;
  if (i < n) o[i] = a[i] * b[i];
}
__global__ void k_mul_inplace(float* a, const float* b, int n) {
  int i = blockIdx.x * 256 + threadIdx.x;
  if (i < n) a[i] *= b[i];
}
__global__ void k_bias_relu(float* x, const float* b, int n) {
  int i = blockIdx.x * 256 + threadIdx.x;
  if (i < n) x[i] = fmaxf(x[i] + b[i & 127], 0.f);
}
__global__ void k_add_relu(const float* a, const float* b, float* o, int n) {
  int i = blockIdx.x * 256 + threadIdx.x;
  if (i < n) o[i] = fmaxf(a[i] + b[i], 0.f);
}

// ---------------------------------------------------------------------------
// GAT attention coefficients: as[n,h] = <h[n,h,:], att_src[h,:]>, ad likewise.
// ---------------------------------------------------------------------------
__global__ void k_gat_coeff(const float* __restrict__ h,
                            const float* __restrict__ att_src,
                            const float* __restrict__ att_dst,
                            float* __restrict__ asb, float* __restrict__ adb,
                            int nn) {
  int i = blockIdx.x * 256 + threadIdx.x;
  if (i >= nn * NHEAD) return;
  int n = i >> 3, hh = i & 7;
  const float* hp = h + (size_t)n * 128 + hh * HDIM;
  float s1 = 0.f, s2 = 0.f;
#pragma unroll
  for (int d = 0; d < HDIM; ++d) {
    float v = hp[d];
    s1 += v * att_src[hh * HDIM + d];
    s2 += v * att_dst[hh * HDIM + d];
  }
  asb[i] = s1;
  adb[i] = s2;
}

// GAT edge pass: logits = leaky_relu(as[src]+ad[dst], 0.2); w = exp(logit);
// segment-sum into ssum[dst].  (Shift-invariance of softmax + tiny logits
// makes the segment-max pass unnecessary.)  e >= E encodes self loops.
__global__ void k_gat_pass1(const int* __restrict__ src, const int* __restrict__ dst,
                            int E_, int Etot,
                            const float* __restrict__ asb, const float* __restrict__ adb,
                            float* __restrict__ wbuf, float* __restrict__ ssum) {
  int i = blockIdx.x * 256 + threadIdx.x;
  if (i >= Etot * NHEAD) return;
  int e = i >> 3, hh = i & 7;
  int s, d;
  if (e < E_) { s = src[e]; d = dst[e]; } else { s = e - E_; d = s; }
  float a = asb[s * NHEAD + hh] + adb[d * NHEAD + hh];
  a = (a > 0.f) ? a : 0.2f * a;
  float w = expf(a);
  wbuf[i] = w;
  atomicAdd(&ssum[d * NHEAD + hh], w);
}

// Transformer edge pass: logits = <q[dst,h,:], k[src,h,:]> / sqrt(16)
__global__ void k_trans_pass1(const int* __restrict__ src, const int* __restrict__ dst,
                              int E_, const float* __restrict__ q, const float* __restrict__ k,
                              float* __restrict__ wbuf, float* __restrict__ ssum) {
  int i = blockIdx.x * 256 + threadIdx.x;
  if (i >= E_ * NHEAD) return;
  int e = i >> 3, hh = i & 7;
  int s = src[e], d = dst[e];
  const float* qp = q + (size_t)d * 128 + hh * HDIM;
  const float* kp = k + (size_t)s * 128 + hh * HDIM;
  float acc = 0.f;
#pragma unroll
  for (int j = 0; j < HDIM; ++j) acc += qp[j] * kp[j];
  float w = expf(acc * 0.25f);
  wbuf[i] = w;
  atomicAdd(&ssum[d * NHEAD + hh], w);
}

// Weighted scatter: out[dst] += val[src] * wbuf[e,h] / (ssum[dst,h] + 1e-16)
// One thread per (edge, 4-channel group): float4 gather + 4 f32 atomics.
__global__ void k_scatter(const int* __restrict__ src, const int* __restrict__ dst,
                          int E_, int Etot,
                          const float* __restrict__ val, const float* __restrict__ wbuf,
                          const float* __restrict__ ssum, float* __restrict__ out) {
  int i = blockIdx.x * 256 + threadIdx.x;
  if (i >= Etot * 32) return;
  int e  = i >> 5;
  int c4 = (i & 31) << 2;      // channel base 0..124
  int s, d;
  if (e < E_) { s = src[e]; d = dst[e]; } else { s = e - E_; d = s; }
  int hh      = c4 >> 4;
  float alpha = wbuf[e * NHEAD + hh] / (ssum[d * NHEAD + hh] + 1e-16f);
  float4 v = *(const float4*)(val + (size_t)s * 128 + c4);
  float* op = out + (size_t)d * 128 + c4;
  atomicAdd(op + 0, v.x * alpha);
  atomicAdd(op + 1, v.y * alpha);
  atomicAdd(op + 2, v.z * alpha);
  atomicAdd(op + 3, v.w * alpha);
}

// Row softmax over 128 channels; one wave32 per node, 4 channels per lane.
__global__ __launch_bounds__(256) void k_softmax128(float* x, int nn) {
  int node = (blockIdx.x * 256 + threadIdx.x) >> 5;
  int lane = threadIdx.x & 31;
  if (node >= nn) return;
  float* row = x + (size_t)node * 128;
  float v[4], mx = -1e30f;
#pragma unroll
  for (int i = 0; i < 4; ++i) { v[i] = row[lane + 32 * i]; mx = fmaxf(mx, v[i]); }
  for (int off = 16; off; off >>= 1) mx = fmaxf(mx, __shfl_xor(mx, off, 32));
  float s = 0.f;
#pragma unroll
  for (int i = 0; i < 4; ++i) { v[i] = expf(v[i] - mx); s += v[i]; }
  for (int off = 16; off; off >>= 1) s += __shfl_xor(s, off, 32);
  float inv = 1.f / s;
#pragma unroll
  for (int i = 0; i < 4; ++i) row[lane + 32 * i] = v[i] * inv;
}

// Gate + residual mix: G = sigmoid(<[z, x3, z-x3], gw> + gb);
// Y = leaky_relu(G*z + (1-G)*x3, 0.25).  One wave32 per node.
__global__ __launch_bounds__(256)
void k_gate_final(const float* __restrict__ z, const float* __restrict__ x3,
                  const float* __restrict__ gw, const float* __restrict__ gb,
                  float* __restrict__ Y, int nn) {
  int node = (blockIdx.x * 256 + threadIdx.x) >> 5;
  int lane = threadIdx.x & 31;
  if (node >= nn) return;
  const float* zp = z + (size_t)node * 128;
  const float* xp = x3 + (size_t)node * 128;
  float zv[4], xv[4], acc = 0.f;
#pragma unroll
  for (int i = 0; i < 4; ++i) {
    int c = lane + 32 * i;
    zv[i] = zp[c];
    xv[i] = xp[c];
    acc += zv[i] * (gw[c] + gw[256 + c]) + xv[i] * (gw[128 + c] - gw[256 + c]);
  }
  for (int off = 16; off; off >>= 1) acc += __shfl_xor(acc, off, 32);
  float G = 1.f / (1.f + expf(-(acc + gb[0])));
#pragma unroll
  for (int i = 0; i < 4; ++i) {
    int c   = lane + 32 * i;
    float o = G * zv[i] + (1.f - G) * xv[i];
    Y[(size_t)node * 128 + c] = (o > 0.f) ? o : 0.25f * o;
  }
}

// Final classifier [128 -> 2] + log_softmax.
__global__ void k_out_logsoftmax(const float* __restrict__ K, const float* __restrict__ Wo,
                                 const float* __restrict__ bo, float* __restrict__ out, int nn) {
  int n = blockIdx.x * 256 + threadIdx.x;
  if (n >= nn) return;
  const float* kp = K + (size_t)n * 128;
  float o0 = bo[0], o1 = bo[1];
#pragma unroll 4
  for (int c = 0; c < 128; ++c) {
    float v = kp[c];
    o0 += v * Wo[c];
    o1 += v * Wo[128 + c];
  }
  float m   = fmaxf(o0, o1);
  float lse = m + logf(expf(o0 - m) + expf(o1 - m));
  out[n * 2 + 0] = o0 - lse;
  out[n * 2 + 1] = o1 - lse;
}

// ---------------------------------------------------------------------------
// Host-side orchestration
// ---------------------------------------------------------------------------
struct GatP { const float *W, *as, *ad, *b; };

static void ctx_layer(const float* X, float* Y,
                      const float* const* fcW, const float* const* fcB,
                      const GatP* gat, const float* const* tr,
                      const float* gateW, const float* gateB,
                      const int* esrc, const int* edst,
                      float* const* B, float* wbuf, float* asb, float* adb, float* ssum,
                      hipStream_t stream) {
  const int GG   = CDIV(N_NODES, 128);
  const int NC   = N_NODES * 128;
  const int Etot = N_EDGES + N_NODES;

  // xs[i] = relu(fc_i(X)) -> B0..B3
  for (int i = 0; i < 4; ++i)
    gemm128_wmma<1><<<GG, 256, 0, stream>>>(X, fcW[i], fcB[i], B[i], N_NODES);

  // Three GATConvs (with self-loops): g0->B5, g1->B0, g2->B1 (reusing dead xs).
  const int gslot[3] = {5, 0, 1};
  for (int i = 0; i < 3; ++i) {
    gemm128_wmma<0><<<GG, 256, 0, stream>>>(B[i], gat[i].W, nullptr, B[4], N_NODES);
    k_gat_coeff<<<CDIV(N_NODES * NHEAD, 256), 256, 0, stream>>>(B[4], gat[i].as, gat[i].ad,
                                                                asb, adb, N_NODES);
    k_fill0<<<CDIV(N_NODES * NHEAD, 256), 256, 0, stream>>>(ssum, N_NODES * NHEAD);
    k_gat_pass1<<<CDIV(Etot * NHEAD, 256), 256, 0, stream>>>(esrc, edst, N_EDGES, Etot,
                                                             asb, adb, wbuf, ssum);
    float* g = B[gslot[i]];
    k_fill0<<<CDIV(NC, 256), 256, 0, stream>>>(g, NC);
    k_scatter<<<CDIV(Etot * 32, 256), 256, 0, stream>>>(esrc, edst, N_EDGES, Etot,
                                                        B[4], wbuf, ssum, g);
    k_bias_relu<<<CDIV(NC, 256), 256, 0, stream>>>(g, gat[i].b, NC);
  }

  // Fc = g0 * g1 -> B2 (xs2 dead)
  k_mul<<<CDIV(NC, 256), 256, 0, stream>>>(B[5], B[0], B[2], NC);

  // TransformerConv: q->B5, k->B6, v->B4, skip->B7
  gemm128_wmma<0><<<GG, 256, 0, stream>>>(B[2], tr[0], tr[1], B[5], N_NODES);
  gemm128_wmma<0><<<GG, 256, 0, stream>>>(B[2], tr[2], tr[3], B[6], N_NODES);
  gemm128_wmma<0><<<GG, 256, 0, stream>>>(B[2], tr[4], tr[5], B[4], N_NODES);
  gemm128_wmma<0><<<GG, 256, 0, stream>>>(B[2], tr[6], tr[7], B[7], N_NODES);

  k_fill0<<<CDIV(N_NODES * NHEAD, 256), 256, 0, stream>>>(ssum, N_NODES * NHEAD);
  k_trans_pass1<<<CDIV(N_EDGES * NHEAD, 256), 256, 0, stream>>>(esrc, edst, N_EDGES,
                                                                B[5], B[6], wbuf, ssum);
  k_fill0<<<CDIV(NC, 256), 256, 0, stream>>>(B[2], NC);                 // Fc dead -> scatter acc
  k_scatter<<<CDIV(N_EDGES * 32, 256), 256, 0, stream>>>(esrc, edst, N_EDGES, N_EDGES,
                                                         B[4], wbuf, ssum, B[2]);
  k_add_relu<<<CDIV(NC, 256), 256, 0, stream>>>(B[2], B[7], B[5], NC);  // m -> B5
  k_softmax128<<<CDIV(N_NODES, 8), 256, 0, stream>>>(B[5], N_NODES);
  k_mul_inplace<<<CDIV(NC, 256), 256, 0, stream>>>(B[5], B[1], NC);     // z = m * g2
  k_gate_final<<<CDIV(N_NODES, 8), 256, 0, stream>>>(B[5], B[3], gateW, gateB, Y, N_NODES);
}

extern "C" void kernel_launch(void* const* d_in, const int* in_sizes, int n_in,
                              void* d_out, int out_size, void* d_ws, size_t ws_size,
                              hipStream_t stream) {
  (void)in_sizes; (void)n_in; (void)out_size; (void)ws_size;

  // d_in layout = setup_inputs() insertion order, params flattened leaf-by-leaf:
  //  0:x 1:edge_index | 2..9: fc[i].{W,b} | 10,11: gate0.{W,b} | 12,13: fcsK.{W,b}
  // 14..21: fc2[i].{W,b} | 22,23: gate1.{W,b} | 24,25: out.{W,b}
  // 26..37: gat0[i].{W,att_src,att_dst,b} | 38..49: gat1[i] | 50..57: trans0 q/k/v/skip {W,b}
  // 58..65: trans1
  auto F = [&](int i) { return (const float*)d_in[i]; };
  const float* x  = F(0);
  const int*   ei = (const int*)d_in[1];
  const int* esrc = ei;
  const int* edst = ei + N_EDGES;

  const float* fc1W[4], *fc1B[4], *fc2W[4], *fc2B[4];
  for (int i = 0; i < 4; ++i) {
    fc1W[i] = F(2 + 2 * i);  fc1B[i] = F(3 + 2 * i);
    fc2W[i] = F(14 + 2 * i); fc2B[i] = F(15 + 2 * i);
  }
  GatP g0[3], g1[3];
  for (int i = 0; i < 3; ++i) {
    g0[i] = {F(26 + 4 * i), F(27 + 4 * i), F(28 + 4 * i), F(29 + 4 * i)};
    g1[i] = {F(38 + 4 * i), F(39 + 4 * i), F(40 + 4 * i), F(41 + 4 * i)};
  }
  const float* t0[8], *t1[8];
  for (int j = 0; j < 8; ++j) { t0[j] = F(50 + j); t1[j] = F(58 + j); }

  // Workspace carve-up (floats)
  float* ws = (float*)d_ws;
  const size_t NB = (size_t)N_NODES * 128;
  float* B[10];
  for (int i = 0; i < 10; ++i) B[i] = ws + i * NB;
  float* wbuf = ws + 10 * NB;                              // (E+N)*8
  float* asb  = wbuf + (size_t)(N_EDGES + N_NODES) * NHEAD;
  float* adb  = asb + (size_t)N_NODES * NHEAD;
  float* ssum = adb + (size_t)N_NODES * NHEAD;

  // Layer 1: x -> B9
  ctx_layer(x, B[9], fc1W, fc1B, g0, t0, F(10), F(11), esrc, edst, B, wbuf, asb, adb, ssum, stream);
  // fcsK: B9 -> B8
  gemm128_wmma<0><<<CDIV(N_NODES, 128), 256, 0, stream>>>(B[9], F(12), F(13), B[8], N_NODES);
  // Layer 2: B8 -> B9
  ctx_layer(B[8], B[9], fc2W, fc2B, g1, t1, F(22), F(23), esrc, edst, B, wbuf, asb, adb, ssum, stream);
  // Classifier + log_softmax -> d_out
  k_out_logsoftmax<<<CDIV(N_NODES, 256), 256, 0, stream>>>(B[9], F(24), F(25),
                                                           (float*)d_out, N_NODES);
}
